// VAE_51539608016
// MI455X (gfx1250) — matrix operand order repaired and checked by reading, not compiled, not run on hardware
//
#include <hip/hip_runtime.h>
#include <hip/hip_bf16.h>
#include <stdint.h>

// ---------------------------------------------------------------------------
// TMC log-space chain, B=128, K=512.  Bandwidth-bound (~402MB reads @ 23.3TB/s
// => ~17us floor).  Kernel 3 streams logP1 in ONE HBM pass via
// GLOBAL_LOAD_ASYNC_TO_LDS_B128 (double-buffered 32x128 chunks, ASYNCcnt
// pipelined) and uses v_wmma_f32_16x16x32_f16 as the K-reduction engine with
// flash-style online (max,sum) rescaling in f32.
// ---------------------------------------------------------------------------

typedef __attribute__((ext_vector_type(16))) _Float16 v16h;
typedef __attribute__((ext_vector_type(8)))  float    v8f;

#define BB 128
#define KK 512
#define LOGK 6.2383246250395075f   // log(512)

// ---------------------------------------------------------------------------
// Kernel 1: column-wise logsumexp of a [B,K,K] tensor -> q[B,K]
// One block handles 256 consecutive columns of one batch; two passes over the
// 512KB slab (max, then exp-sum) -- 2nd pass hits L2/WGP$.
// ---------------------------------------------------------------------------
__global__ void k_col_lse(const float* __restrict__ Q, float* __restrict__ qout) {
    const int idx = blockIdx.x;                 // 0..255
    const int b   = idx >> 1;
    const int j   = ((idx & 1) << 8) + threadIdx.x;
    const float* col = Q + (size_t)b * KK * KK + j;

    float m = -__builtin_inff();
    for (int i = 0; i < KK; ++i)
        m = fmaxf(m, col[(size_t)i * KK]);
    float s = 0.f;
    for (int i = 0; i < KK; ++i)
        s += __expf(col[(size_t)i * KK] - m);
    qout[b * KK + j] = m + __logf(s) - LOGK;
}

// ---------------------------------------------------------------------------
// Kernel 2: v = logP0 - q0, x[b] = max_j v, a_h[b,j] = (f16) exp(v - x)
// ---------------------------------------------------------------------------
__global__ void k_avec(const float* __restrict__ P0, const float* __restrict__ q0,
                       _Float16* __restrict__ a_h, float* __restrict__ xb) {
    const int b = blockIdx.x, tid = threadIdx.x;
    __shared__ float sm[256];

    float v1 = P0[b * KK + tid]       - q0[b * KK + tid];
    float v2 = P0[b * KK + tid + 256] - q0[b * KK + tid + 256];
    sm[tid] = fmaxf(v1, v2);
    __syncthreads();
    for (int s = 128; s > 0; s >>= 1) {
        if (tid < s) sm[tid] = fmaxf(sm[tid], sm[tid + s]);
        __syncthreads();
    }
    const float x = sm[0];
    a_h[b * KK + tid]       = (_Float16)__expf(v1 - x);
    a_h[b * KK + tid + 256] = (_Float16)__expf(v2 - x);
    if (tid == 0) xb[b] = x;
}

// ---------------------------------------------------------------------------
// Kernel 3: w[j] = x + (M[j]-q1[j]) + log(sum_i a[i]*exp(P1[i,j]-M[j])) - logK
//
// One workgroup (8 wave32) per (batch, 128-column stripe).  logP1 is streamed
// once from HBM via async B128 copies into a double-buffered 32x128 LDS chunk;
// each wave owns a 16-column tile and runs an online (m,s) recurrence:
//   per chunk: m_c = chunk column max (from LDS)
//              s_c = WMMA( a_f16 replicated rows ,  exp(p - m_c) )   [f32 acc]
//              s   = s*exp(m - m') + s_c*exp(m_c - m');  m = m'
// This is exact (m ends as the true column max) and takes ONE HBM pass.
// ---------------------------------------------------------------------------
__global__ void k_wmma_step(const float* __restrict__ P1, const float* __restrict__ q1,
                            const _Float16* __restrict__ a_h, const float* __restrict__ xb,
                            float* __restrict__ wout) {
    const int b      = blockIdx.x >> 2;
    const int stripe = blockIdx.x & 3;
    const int j0     = stripe * 128;          // stripe column base
    const int tid    = threadIdx.x;
    const int lane   = tid & 31;
    const int wave   = tid >> 5;
    const int hi     = lane >> 4;             // half-wave id
    const int li     = lane & 15;

    __shared__ float tile[2][32][128];        // 32 KB double buffer

    const float* P1b = P1 + (size_t)b * KK * KK;
    const unsigned tbase = (unsigned)(uintptr_t)&tile[0][0][0];

    // issue one 32x128 f32 chunk: 32 rows x 512B; wave covers rows wave+8i,
    // each lane moves 16B -> coalesced GLOBAL_LOAD_ASYNC_TO_LDS_B128.
    auto issue_chunk = [&](int buf, int chunk) {
#pragma unroll
        for (int i = 0; i < 4; ++i) {
            const int r = wave + 8 * i;
            const unsigned ldsa = tbase + (unsigned)(buf * 16384 + r * 512 + lane * 16);
            const uint64_t ga   = (uint64_t)(uintptr_t)(P1b + (size_t)(chunk * 32 + r) * KK + j0)
                                + (unsigned)(lane * 16);
            asm volatile("global_load_async_to_lds_b128 %0, %1, off"
                         :: "v"(ldsa), "v"(ga) : "memory");
        }
    };

    const int   cw  = wave * 16 + li;         // LDS column this lane owns
    const int   col = j0 + cw;                // global column
    const float xbv = xb[b];

    issue_chunk(0, 0);

    float m_run = -__builtin_inff();
    float s_run = 0.f;

    for (int c = 0; c < 16; ++c) {
        const int buf = c & 1;
        if (c + 1 < 16) {
            issue_chunk(buf ^ 1, c + 1);
            asm volatile("s_wait_asynccnt 0x4" ::: "memory");  // chunk c landed
        } else {
            asm volatile("s_wait_asynccnt 0x0" ::: "memory");
        }
        __syncthreads();                      // all waves' rows of chunk c visible

        // chunk-local column max
        float m_c = -__builtin_inff();
#pragma unroll
        for (int r = 0; r < 32; ++r)
            m_c = fmaxf(m_c, tile[buf][r][cw]);

        // fragments: A = `a` slice replicated across 16 rows (same K-map as B),
        //            B = exp(p - m_c), lanes 0-15 K=0..15, lanes 16-31 K=16..31
        v16h af, bf;
        const _Float16* ap = a_h + b * KK + c * 32 + hi * 16;
#pragma unroll
        for (int e = 0; e < 16; ++e) {
            const int ka = e;
            af[e] = ap[ka];
            bf[e] = (_Float16)__expf(tile[buf][hi * 16 + e][cw] - m_c);
        }
        v8f pc = {};
        pc = __builtin_amdgcn_wmma_f32_16x16x32_f16(
                 /*neg_a=*/false, af, /*neg_b=*/false, bf,
                 /*c_mod=*/(short)0, pc, /*reuse_a=*/false, /*reuse_b=*/false);
        const float s_c = pc[0];              // all D rows identical

        // online rescale combine (f32)
        const float mn = fmaxf(m_run, m_c);
        s_run = s_run * __expf(m_run - mn) + s_c * __expf(m_c - mn);
        m_run = mn;

        __syncthreads();                      // buf free for chunk c+2
    }

    if (lane < 16) {
        wout[b * KK + col] = xbv + (m_run - q1[b * KK + col]) + __logf(s_run) - LOGK;
    }
}

// ---------------------------------------------------------------------------
// Kernel 4: final log-space dot with logP2 per batch, sum & negate -> scalar.
// ---------------------------------------------------------------------------
__global__ void k_final(const float* __restrict__ w, const float* __restrict__ p2,
                        float* __restrict__ out) {
    __shared__ float sm[256];
    __shared__ float sm2[256];
    const int tid = threadIdx.x;
    float total = 0.f;

    for (int b = 0; b < BB; ++b) {
        const float w1 = w[b * KK + tid],  w2 = w[b * KK + tid + 256];
        const float pA = p2[b * KK + tid], pB = p2[b * KK + tid + 256];
        sm[tid]  = fmaxf(w1, w2);
        sm2[tid] = fmaxf(pA, pB);
        __syncthreads();
        for (int s = 128; s > 0; s >>= 1) {
            if (tid < s) {
                sm[tid]  = fmaxf(sm[tid],  sm[tid + s]);
                sm2[tid] = fmaxf(sm2[tid], sm2[tid + s]);
            }
            __syncthreads();
        }
        const float x2 = sm[0], y2 = sm2[0];
        __syncthreads();
        sm[tid] = __expf(w1 - x2) * __expf(pA - y2) + __expf(w2 - x2) * __expf(pB - y2);
        __syncthreads();
        for (int s = 128; s > 0; s >>= 1) {
            if (tid < s) sm[tid] += sm[tid + s];
            __syncthreads();
        }
        if (tid == 0) total += x2 + y2 + __logf(sm[0]) - LOGK;
        __syncthreads();
    }
    if (tid == 0) *out = -total;
}

// ---------------------------------------------------------------------------
extern "C" void kernel_launch(void* const* d_in, const int* in_sizes, int n_in,
                              void* d_out, int out_size, void* d_ws, size_t ws_size,
                              hipStream_t stream) {
    const float* logQ0 = (const float*)d_in[0];   // [B,K,K]
    const float* logQ1 = (const float*)d_in[1];   // [B,K,K]
    const float* logP0 = (const float*)d_in[2];   // [B,1,K]
    const float* logP1 = (const float*)d_in[3];   // [B,K,K]
    const float* logP2 = (const float*)d_in[4];   // [B,K,1]
    float* out = (float*)d_out;

    // workspace carve-out (< 1 MB)
    char* ws = (char*)d_ws;
    float*     q0  = (float*)(ws);                          // B*K f32 = 256KB
    float*     q1  = (float*)(ws + 262144);                 // B*K f32 = 256KB
    float*     xb  = (float*)(ws + 524288);                 // B f32
    _Float16*  a_h = (_Float16*)(ws + 528384);              // B*K f16 = 128KB
    float*     wv  = (float*)(ws + 659456);                 // B*K f32 = 256KB

    // q0 <- col-LSE(logQ1), q1 <- col-LSE(logQ0)   (note the swap per reference)
    k_col_lse<<<BB * 2, 256, 0, stream>>>(logQ1, q0);
    k_col_lse<<<BB * 2, 256, 0, stream>>>(logQ0, q1);
    k_avec<<<BB, 256, 0, stream>>>(logP0, q0, a_h, xb);
    k_wmma_step<<<BB * 4, 256, 0, stream>>>(logP1, q1, a_h, xb, wv);
    k_final<<<1, 256, 0, stream>>>(wv, logP2, out);
}
